// Attention_63436666961990
// MI455X (gfx1250) — matrix operand order repaired and checked by reading, not compiled, not run on hardware
//
#include <hip/hip_runtime.h>
#include <hip/hip_bf16.h>

typedef __bf16 bf16_t;
typedef __attribute__((ext_vector_type(16))) __bf16 v16bf;
typedef __attribute__((ext_vector_type(8)))  __bf16 v8bf;
typedef __attribute__((ext_vector_type(8)))  float  v8f;

#define E_DIM 768
#define QKV_DIM 2304
#define NHEAD 12
#define HDIM 64
#define TOK 197
#define BATCH 16
#define NFRM 12
#define NROWS (BATCH*NFRM*TOK)   /* 37824 */

static __device__ __forceinline__ bf16_t f2bf(float f) {
  union { float f; unsigned u; } a; a.f = f;
  unsigned r = a.u + 0x7FFFu + ((a.u >> 16) & 1u);
  union { unsigned short s; bf16_t b; } o; o.s = (unsigned short)(r >> 16);
  return o.b;
}

static __device__ __forceinline__ v8f vzero8() {
  v8f z;
#pragma unroll
  for (int i = 0; i < 8; ++i) z[i] = 0.0f;
  return z;
}

// A-fragment (16x32 bf16, M = lane%16): K = 16 contiguous values starting at 16*half
static __device__ __forceinline__ v16bf load_afrag(const bf16_t* rowbase, int lh) {
  return *(const v16bf*)(rowbase + (lh ? 16 : 0));
}
// B-fragment (32x16 bf16) from N-major (B^T row-major) storage: N = lane%16,
// K = 16 contiguous values starting at 16*half
static __device__ __forceinline__ v16bf load_bfrag(const bf16_t* rowbase, int lh) {
  return *(const v16bf*)(rowbase + (lh ? 16 : 0));
}
static __device__ __forceinline__ v8f wmma_bf16(v16bf a, v16bf b, v8f c) {
  return __builtin_amdgcn_wmma_f32_16x16x32_bf16(false, a, false, b, (short)0, c, false, false);
}

// async global->LDS copy of 16 bytes per lane (CDNA5, ASYNCcnt-tracked)
static __device__ __forceinline__ void async_b128(unsigned lds_off, const void* gaddr) {
  asm volatile("global_load_async_to_lds_b128 %0, %1, off"
               :: "v"(lds_off), "v"((unsigned long long)(size_t)gaddr)
               : "memory");
}

// ---------------- weight prep ----------------
__global__ void prep_wqkv(const float* __restrict__ W, const float* __restrict__ la,
                          const float* __restrict__ lb, bf16_t* __restrict__ out) {
  int idx = blockIdx.x * 256 + threadIdx.x;
  if (idx >= QKV_DIM * E_DIM) return;
  int i = idx / E_DIM, j = idx - i * E_DIM;
  float acc = W[idx];
#pragma unroll
  for (int r = 0; r < 8; ++r) acc += lb[i * 8 + r] * la[r * E_DIM + j];
  out[idx] = f2bf(acc);
}

__global__ void cvt_f32_bf16(const float* __restrict__ in, bf16_t* __restrict__ out, long n) {
  long i = (long)blockIdx.x * blockDim.x + threadIdx.x;
  if (i < n) out[i] = f2bf(in[i]);
}

// ---------------- WMMA GEMM, async-LDS double-buffered ----------------
// C[M,N] = A(M,K) @ W(N,K)^T + bias.  Block tile 64x128, BK=32, 8 waves (2x4), wave tile 32x32.
__global__ __launch_bounds__(256) void gemm_bf16_wmma(
    const bf16_t* __restrict__ A, const bf16_t* __restrict__ W,
    const float* __restrict__ bias, bf16_t* __restrict__ Cb,
    float* __restrict__ Cf, int N, int K) {
  __shared__ bf16_t As[2][64 * 32] __attribute__((aligned(128)));
  __shared__ bf16_t Bs[2][128 * 32] __attribute__((aligned(128)));
  const int tid = threadIdx.x;
  const int lane = tid & 31, wave = tid >> 5;
  const int lh = lane >> 4, l16 = lane & 15;
  const int m_off = (wave >> 2) * 32, n_off = (wave & 3) * 32;
  const int bm0 = blockIdx.x * 64, bn0 = blockIdx.y * 128;

  const unsigned aBase = (unsigned)(size_t)(void*)&As[0][0];
  const unsigned bBase = (unsigned)(size_t)(void*)&Bs[0][0];
  // staging geometry: rows of 32 bf16 (64B) split into 4 x 16B segments
  const int arow = tid >> 2, aseg = tid & 3;         // 64 rows x 4 segs = 256 chunks
  const int brow1 = (tid >> 2) + 64;                 // B chunk 2: rows 64..127

  v8f acc[4];
#pragma unroll
  for (int c = 0; c < 4; ++c) acc[c] = vzero8();

  auto issue = [&](int ks, int buf) {
    const int k0 = ks * 32;
    async_b128(aBase + (unsigned)(buf * 64 * 32 * 2 + arow * 64 + aseg * 16),
               A + (size_t)(bm0 + arow) * K + k0 + aseg * 8);
    async_b128(bBase + (unsigned)(buf * 128 * 32 * 2 + arow * 64 + aseg * 16),
               W + (size_t)(bn0 + arow) * K + k0 + aseg * 8);
    async_b128(bBase + (unsigned)(buf * 128 * 32 * 2 + brow1 * 64 + aseg * 16),
               W + (size_t)(bn0 + brow1) * K + k0 + aseg * 8);
  };

  const int NKS = K / 32;
  issue(0, 0);
  for (int ks = 0; ks < NKS; ++ks) {
    const int buf = ks & 1;
    if (ks + 1 < NKS) {
      issue(ks + 1, buf ^ 1);
      asm volatile("s_wait_asynccnt 0x3" ::: "memory");  // previous stage (3 ops) done
    } else {
      asm volatile("s_wait_asynccnt 0x0" ::: "memory");
    }
    __syncthreads();  // stage ks visible to all waves
    const bf16_t* ab = &As[buf][0];
    const bf16_t* bb = &Bs[buf][0];
    v16bf a0 = load_afrag(ab + (m_off + l16) * 32, lh);
    v16bf a1 = load_afrag(ab + (m_off + 16 + l16) * 32, lh);
    v16bf b0 = load_bfrag(bb + (n_off + l16) * 32, lh);
    v16bf b1 = load_bfrag(bb + (n_off + 16 + l16) * 32, lh);
    acc[0] = wmma_bf16(a0, b0, acc[0]);
    acc[1] = wmma_bf16(a0, b1, acc[1]);
    acc[2] = wmma_bf16(a1, b0, acc[2]);
    acc[3] = wmma_bf16(a1, b1, acc[3]);
    __syncthreads();  // all reads of buf done before it is refilled
  }

#pragma unroll
  for (int ai = 0; ai < 2; ++ai)
#pragma unroll
    for (int bi2 = 0; bi2 < 2; ++bi2) {
      v8f a = acc[ai * 2 + bi2];
      int col = bn0 + n_off + bi2 * 16 + l16;
      float bv = bias ? bias[col] : 0.0f;
#pragma unroll
      for (int g = 0; g < 8; ++g) {
        int row = bm0 + m_off + ai * 16 + g + lh * 8;
        float v = a[g] + bv;
        if (Cb) Cb[(size_t)row * N + col] = f2bf(v);
        else    Cf[(size_t)row * N + col] = v;
      }
    }
}

// ---------------- flash attention (mode 0: per-frame, mode 1: cross-frame) ----------------
__global__ __launch_bounds__(256) void attn_kernel(const bf16_t* __restrict__ qkv,
                                                   bf16_t* __restrict__ outb, int mode) {
  __shared__ bf16_t Vt[HDIM * 32] __attribute__((aligned(64)));      // [d][key]
  __shared__ bf16_t Pbuf[8][16 * 32] __attribute__((aligned(64)));   // per-wave P tile
  const int tid = threadIdx.x, lane = tid & 31, wave = tid >> 5;
  const int lh = lane >> 4, l16 = lane & 15;
  int h, qrow0, qstride, nq, krow0, nk, orow0;
  if (mode == 0) {
    int bb = blockIdx.x / NHEAD; h = blockIdx.x % NHEAD;
    qrow0 = bb * TOK; qstride = 1; nq = TOK;
    krow0 = qrow0; nk = TOK; orow0 = bb * TOK;
  } else {
    int b = blockIdx.x / NHEAD; h = blockIdx.x % NHEAD;
    qrow0 = b * NFRM * TOK; qstride = TOK; nq = NFRM;
    krow0 = qrow0; nk = NFRM * TOK; orow0 = b * NFRM;
  }
  const int qcol = h * HDIM, kcol = E_DIM + h * HDIM, vcol = 2 * E_DIM + h * HDIM;
  const float scaling = 0.125f;  // 64^-0.5
  const int nqt = (nq + 15) >> 4;

  int tiles[2]; int nmy = 0;
  for (int t = wave; t < nqt && nmy < 2; t += 8) tiles[nmy++] = t;

  v16bf qf[2][2]; v8f O[2][4]; float rm[2][8], rl[2][8];
#pragma unroll
  for (int qi = 0; qi < 2; ++qi) if (qi < nmy) {
    int m = tiles[qi] * 16 + l16; if (m > nq - 1) m = nq - 1;
    const bf16_t* qb = qkv + ((size_t)qrow0 + (size_t)m * qstride) * QKV_DIM + qcol;
    qf[qi][0] = load_afrag(qb, lh);
    qf[qi][1] = load_afrag(qb + 32, lh);
#pragma unroll
    for (int c = 0; c < 4; ++c) O[qi][c] = vzero8();
#pragma unroll
    for (int g = 0; g < 8; ++g) { rm[qi][g] = -3.0e38f; rl[qi][g] = 0.0f; }
  }

  const int nkb = (nk + 31) / 32;
  for (int kb = 0; kb < nkb; ++kb) {
    __syncthreads();  // protect Vt readers of previous block
    {   // stage V^T tile: Vt[d][key]
      int key = tid & 31, d0 = (tid >> 5) * 8;
      int kidx = kb * 32 + key;
      if (kidx < nk) {
        v8bf vv = *(const v8bf*)(qkv + (size_t)(krow0 + kidx) * QKV_DIM + vcol + d0);
#pragma unroll
        for (int i = 0; i < 8; ++i) Vt[(d0 + i) * 32 + key] = vv[i];
      } else {
#pragma unroll
        for (int i = 0; i < 8; ++i) Vt[(d0 + i) * 32 + key] = f2bf(0.0f);
      }
    }
    __syncthreads();

#pragma unroll
    for (int qi = 0; qi < 2; ++qi) if (qi < nmy) {
      v8f S0 = vzero8(), S1 = vzero8();
      {
        int k0i = kb * 32 + l16;       int kc0 = k0i < nk ? k0i : nk - 1;
        int k1i = kb * 32 + 16 + l16;  int kc1 = k1i < nk ? k1i : nk - 1;
        const bf16_t* kr0 = qkv + (size_t)(krow0 + kc0) * QKV_DIM + kcol;
        const bf16_t* kr1 = qkv + (size_t)(krow0 + kc1) * QKV_DIM + kcol;
        S0 = wmma_bf16(qf[qi][0], load_bfrag(kr0, lh), S0);
        S0 = wmma_bf16(qf[qi][1], load_bfrag(kr0 + 32, lh), S0);
        S1 = wmma_bf16(qf[qi][0], load_bfrag(kr1, lh), S1);
        S1 = wmma_bf16(qf[qi][1], load_bfrag(kr1 + 32, lh), S1);
      }
      bool v0 = (kb * 32 + l16) < nk, v1 = (kb * 32 + 16 + l16) < nk;
#pragma unroll
      for (int g = 0; g < 8; ++g) {
        S0[g] = v0 ? S0[g] * scaling : -3.0e38f;
        S1[g] = v1 ? S1[g] * scaling : -3.0e38f;
      }
      float bm[8];
#pragma unroll
      for (int g = 0; g < 8; ++g) bm[g] = fmaxf(S0[g], S1[g]);
#pragma unroll
      for (int off = 1; off < 16; off <<= 1)
#pragma unroll
        for (int g = 0; g < 8; ++g) bm[g] = fmaxf(bm[g], __shfl_xor(bm[g], off, 32));
      float rs[8];
#pragma unroll
      for (int g = 0; g < 8; ++g) {
        float nm = fmaxf(rm[qi][g], bm[g]);
        float cf = __expf(rm[qi][g] - nm);
        rm[qi][g] = nm;
        S0[g] = __expf(S0[g] - nm);
        S1[g] = __expf(S1[g] - nm);
        rs[g] = S0[g] + S1[g];
        rl[qi][g] *= cf;
#pragma unroll
        for (int c = 0; c < 4; ++c) O[qi][c][g] *= cf;
      }
#pragma unroll
      for (int off = 1; off < 16; off <<= 1)
#pragma unroll
        for (int g = 0; g < 8; ++g) rs[g] += __shfl_xor(rs[g], off, 32);
#pragma unroll
      for (int g = 0; g < 8; ++g) rl[qi][g] += rs[g];

      // P (C-layout) -> bf16 A-layout via wave-private LDS bounce
      bf16_t* pb = &Pbuf[wave][0];
#pragma unroll
      for (int g = 0; g < 8; ++g) {
        pb[(g + lh * 8) * 32 + l16]      = f2bf(S0[g]);
        pb[(g + lh * 8) * 32 + 16 + l16] = f2bf(S1[g]);
      }
      v16bf pf = load_afrag(pb + l16 * 32, lh);
#pragma unroll
      for (int c = 0; c < 4; ++c)
        O[qi][c] = wmma_bf16(pf, load_bfrag(&Vt[(c * 16 + l16) * 32], lh), O[qi][c]);
    }
  }

#pragma unroll
  for (int qi = 0; qi < 2; ++qi) if (qi < nmy) {
#pragma unroll
    for (int g = 0; g < 8; ++g) {
      float inv = 1.0f / rl[qi][g];
      int row = tiles[qi] * 16 + g + lh * 8;
      if (row < nq) {
#pragma unroll
        for (int c = 0; c < 4; ++c)
          outb[(size_t)(orow0 + row) * E_DIM + h * HDIM + c * 16 + l16] =
              f2bf(O[qi][c][g] * inv);
      }
    }
  }
}

// ---------------- x1 = ctx1 @ Wout^T + bias (192x768, small) ----------------
__global__ void x1_proj(const bf16_t* __restrict__ ctx1, const bf16_t* __restrict__ wout,
                        const float* __restrict__ bias, float* __restrict__ x1) {
  int row = blockIdx.x;
  for (int col = threadIdx.x; col < E_DIM; col += 256) {
    float acc = bias[col];
    for (int k = 0; k < E_DIM; ++k)
      acc += (float)ctx1[(size_t)row * E_DIM + k] * (float)wout[(size_t)col * E_DIM + k];
    x1[(size_t)row * E_DIM + col] = acc;
  }
}

// ---------------- token-0 adapter + x1 fixup ----------------
__global__ void finalize_tok0(float* __restrict__ out, const float* __restrict__ x1,
                              const float* __restrict__ dw, const float* __restrict__ db,
                              const float* __restrict__ uw, const float* __restrict__ ub) {
  int bf = blockIdx.x;  // bsz index b*12+f; token 0 row
  __shared__ float x0[E_DIM];
  __shared__ float gact[8];
  float* row = out + (size_t)bf * TOK * E_DIM;
  for (int j = threadIdx.x; j < E_DIM; j += 256) x0[j] = row[j];
  __syncthreads();
  if (threadIdx.x < 8) {
    float a = db[threadIdx.x];
    for (int k = 0; k < E_DIM; ++k) a += dw[threadIdx.x * E_DIM + k] * x0[k];
    gact[threadIdx.x] = a / (1.0f + __expf(-1.702f * a));  // quick_gelu
  }
  __syncthreads();
  for (int j = threadIdx.x; j < E_DIM; j += 256) {
    float a = ub[j];
#pragma unroll
    for (int r = 0; r < 8; ++r) a += gact[r] * uw[j * 8 + r];
    row[j] = a + x1[(size_t)bf * E_DIM + j];
  }
}

extern "C" void kernel_launch(void* const* d_in, const int* in_sizes, int n_in,
                              void* d_out, int out_size, void* d_ws, size_t ws_size,
                              hipStream_t stream) {
  (void)in_sizes; (void)n_in; (void)out_size; (void)ws_size;
  const float* x  = (const float*)d_in[0];
  const float* wi = (const float*)d_in[1];
  const float* bi = (const float*)d_in[2];
  const float* wo = (const float*)d_in[3];
  const float* bo = (const float*)d_in[4];
  const float* la = (const float*)d_in[5];
  const float* lb = (const float*)d_in[6];
  const float* dw = (const float*)d_in[7];
  const float* db = (const float*)d_in[8];
  const float* uw = (const float*)d_in[9];
  const float* ub = (const float*)d_in[10];
  float* out = (float*)d_out;

  char* ws = (char*)d_ws;
  size_t off = 0;
  auto alloc = [&](size_t bytes) {
    void* p = (void*)(ws + off);
    off += (bytes + 255) & ~(size_t)255;
    return p;
  };
  bf16_t* xb   = (bf16_t*)alloc((size_t)NROWS * E_DIM * 2);
  bf16_t* wqkv = (bf16_t*)alloc((size_t)QKV_DIM * E_DIM * 2);
  bf16_t* wob  = (bf16_t*)alloc((size_t)E_DIM * E_DIM * 2);
  bf16_t* qkv  = (bf16_t*)alloc((size_t)NROWS * QKV_DIM * 2);
  bf16_t* ctx  = (bf16_t*)alloc((size_t)NROWS * E_DIM * 2);
  bf16_t* ctx1 = (bf16_t*)alloc((size_t)BATCH * NFRM * E_DIM * 2);
  float*  x1   = (float*)alloc((size_t)BATCH * NFRM * E_DIM * 4);

  prep_wqkv<<<(QKV_DIM * E_DIM + 255) / 256, 256, 0, stream>>>(wi, la, lb, wqkv);
  cvt_f32_bf16<<<((long)NROWS * E_DIM + 255) / 256, 256, 0, stream>>>(x, xb, (long)NROWS * E_DIM);
  cvt_f32_bf16<<<(E_DIM * E_DIM + 255) / 256, 256, 0, stream>>>(wo, wob, (long)E_DIM * E_DIM);

  // qkv = x @ Weff^T + bias  (M=37824, N=2304, K=768); block tile 64x128
  gemm_bf16_wmma<<<dim3(NROWS / 64, QKV_DIM / 128), 256, 0, stream>>>(
      xb, wqkv, bi, qkv, nullptr, QKV_DIM, E_DIM);

  // per-frame attention -> ctx (bsz*197 rows)
  attn_kernel<<<BATCH * NFRM * NHEAD, 256, 0, stream>>>(qkv, ctx, 0);
  // cross-frame attention -> ctx1 (b*12 rows)
  attn_kernel<<<BATCH * NHEAD, 256, 0, stream>>>(qkv, ctx1, 1);

  // out = ctx @ Wout^T + bias -> d_out (fp32)
  gemm_bf16_wmma<<<dim3(NROWS / 64, E_DIM / 128), 256, 0, stream>>>(
      ctx, wob, bo, nullptr, out, E_DIM, E_DIM);

  // x1 = ctx1 @ Wout^T + bias ; then token-0 adapter fixup
  x1_proj<<<BATCH * NFRM, 256, 0, stream>>>(ctx1, wob, bo, x1);
  finalize_tok0<<<BATCH * NFRM, 256, 0, stream>>>(out, x1, dw, db, uw, ub);
}
